// VanillaMPNN_89876485636142
// MI455X (gfx1250) — compile-verified, matrix operand order checked by reading, not run on hardware
//
#include <hip/hip_runtime.h>
#include <hip/hip_bf16.h>

// ---------------------------------------------------------------------------
// MPNN on gfx1250: bf16 WMMA (v_wmma_f32_16x16x32_bf16) for edge/node MLPs,
// fp32 state + fp32 atomics for segment sums. L2-resident feature tables.
// ROWSTRIDE=36 -> 16B-aligned LDS rows: fragment loads are ds_load_b128 and
// stage-1 row writes are ds_store_b128 (vs b32/2addr with odd stride).
// ---------------------------------------------------------------------------

typedef __attribute__((ext_vector_type(16))) __bf16   v16bf;
typedef __attribute__((ext_vector_type(8)))  float    v8f;
typedef __attribute__((ext_vector_type(8)))  unsigned v8u;

#define D          48
#define ROWSTRIDE  36            // dwords per LDS row (24 data + 8 zero-pad + 4) ; mult of 4
#define WG_ROWS    256           // edges (or nodes) per workgroup
#define THREADS    256           // 8 waves of 32
#define TILES_PER_WAVE 2         // 256 rows / 16 rows-per-tile / 8 waves

__device__ __forceinline__ unsigned bf16_rne(float f) {
    unsigned u = __float_as_uint(f);
    return (u + 0x7fffu + ((u >> 16) & 1u)) >> 16;
}
__device__ __forceinline__ unsigned pack2(float a, float b) {
    return bf16_rne(a) | (bf16_rne(b) << 16);
}

// Build a 16x32 bf16 A-fragment for this lane from an LDS row of bf16 pairs.
// A layout (ISA 7.12.2): lanes 0-15 row M=lane hold K 0..7 & 16..23,
// lanes 16-31 row M=lane-16 hold K 8..15 & 24..31. kc selects K-chunk (0/1).
// Row base is 16B-aligned and (kc*16+half*4) is a multiple of 4 -> b128 loads.
__device__ __forceinline__ v16bf frag_from_lds(const unsigned* row, int kc, int half) {
    const uint4* q = (const uint4*)(row + kc * 16 + half * 4);
    uint4 lo = q[0];
    uint4 hi = q[2];                 // +8 dwords
    v8u d;
    d[0] = lo.x; d[1] = lo.y; d[2] = lo.z; d[3] = lo.w;
    d[4] = hi.x; d[5] = hi.y; d[6] = hi.z; d[7] = hi.w;
    return __builtin_bit_cast(v16bf, d);
}

__device__ __forceinline__ v16bf load_wfrag(const unsigned* base, int f, int lane) {
    return __builtin_bit_cast(v16bf, ((const v8u*)base)[f * 32 + lane]);
}

__device__ __forceinline__ void atomAddF(float* p, float v) {
    unsafeAtomicAdd(p, v);   // global_atomic_add_f32
}

// ---------------------------------------------------------------------------
// Pre-swizzle one 48x48 weight matrix into 6 per-lane B fragments
// (kc in {0,1} x nt in {0,1,2}), K padded to 64 with zeros, bf16.
// Fragment layout mirrors the documented A layout (N across lanes, K packed).
// ---------------------------------------------------------------------------
__global__ void prep_frags(const float* __restrict__ msg_w1, const float* __restrict__ msg_w2,
                           const float* __restrict__ upd_w1, const float* __restrict__ upd_w2,
                           unsigned* __restrict__ frags) {
    int mat   = blockIdx.x;          // 0..23
    int grp   = mat / 6;
    int layer = mat % 6;
    const float* W = (grp == 0) ? msg_w1 : (grp == 1) ? msg_w2 : (grp == 2) ? upd_w1 : upd_w2;
    W += (size_t)layer * D * D;

    int t    = threadIdx.x;          // 192 = 6 frags x 32 lanes
    int f    = t >> 5, lane = t & 31;
    int kc   = f / 3,  nt   = f % 3;
    int half = lane >> 4;
    int n    = nt * 16 + (lane & 15);
    unsigned* dstp = frags + (size_t)mat * 1536 + f * 256 + lane * 8;
#pragma unroll
    for (int p = 0; p < 8; ++p) {
        int k0 = ((p < 4) ? 2 * p : 8 + 2 * p) + half * 8 + kc * 32;
        float v0 = 0.f, v1 = 0.f;
        if (k0 < D) { v0 = W[k0 * D + n]; v1 = W[(k0 + 1) * D + n]; }
        dstp[p] = pack2(v0, v1);
    }
}

__global__ void embed_kernel(const int* __restrict__ an, const float* __restrict__ emb,
                             float* __restrict__ h, float* __restrict__ agg, int total) {
    int i = blockIdx.x * THREADS + threadIdx.x;
    if (i < total) {
        int n = i / D, d = i % D;
        h[i]   = emb[an[n] * D + d];
        agg[i] = 0.f;
    }
}

__global__ void zero_out_kernel(float* __restrict__ out, int n) {
    int i = blockIdx.x * THREADS + threadIdx.x;
    if (i < n) out[i] = 0.f;
}

// ---------------------------------------------------------------------------
// Edge kernel: x = h[src]*h[dst]; m = relu(x@W1+b1)@W2+b2; agg[dst] += m
// ---------------------------------------------------------------------------
__global__ __launch_bounds__(THREADS) void edge_kernel(
    const float* __restrict__ h, float* __restrict__ agg,
    const int* __restrict__ src, const int* __restrict__ dst,
    const unsigned* __restrict__ w1f, const float* __restrict__ b1,
    const unsigned* __restrict__ w2f, const float* __restrict__ b2, int E) {

    __shared__ unsigned xs[WG_ROWS * ROWSTRIDE];
    int eBase = blockIdx.x * WG_ROWS;

    // ---- Stage 1: gather + multiply -> bf16 rows in LDS (b128 stores) -----
    {
        int r = threadIdx.x;
        int e = eBase + r;
        uint4* row4 = (uint4*)&xs[r * ROWSTRIDE];
        if (e < E) {
            const float4* hs = (const float4*)(h + (size_t)src[e] * D);
            const float4* hd = (const float4*)(h + (size_t)dst[e] * D);
#pragma unroll
            for (int j = 0; j < 6; ++j) {
                float4 a0 = hs[2 * j], a1 = hs[2 * j + 1];
                float4 b0 = hd[2 * j], b1_ = hd[2 * j + 1];
                uint4 o;
                o.x = pack2(a0.x * b0.x, a0.y * b0.y);
                o.y = pack2(a0.z * b0.z, a0.w * b0.w);
                o.z = pack2(a1.x * b1_.x, a1.y * b1_.y);
                o.w = pack2(a1.z * b1_.z, a1.w * b1_.w);
                row4[j] = o;
            }
        } else {
#pragma unroll
            for (int j = 0; j < 6; ++j) row4[j] = make_uint4(0u, 0u, 0u, 0u);
        }
        row4[6] = make_uint4(0u, 0u, 0u, 0u);   // K 48..63 zero pad
        row4[7] = make_uint4(0u, 0u, 0u, 0u);
    }
    __syncthreads();

    int lane = threadIdx.x & 31;
    int wave = threadIdx.x >> 5;
    int half = lane >> 4, ln = lane & 15;

    v16bf w1[6], w2[6];
#pragma unroll
    for (int f = 0; f < 6; ++f) { w1[f] = load_wfrag(w1f, f, lane); w2[f] = load_wfrag(w2f, f, lane); }
    float b1v[3], b2v[3];
#pragma unroll
    for (int nt = 0; nt < 3; ++nt) { b1v[nt] = b1[nt * 16 + ln]; b2v[nt] = b2[nt * 16 + ln]; }

    unsigned short* xs16 = (unsigned short*)xs;

    for (int t = 0; t < TILES_PER_WAVE; ++t) {
        int rowBase = (wave * TILES_PER_WAVE + t) * 16;
        const unsigned* arow = &xs[(rowBase + ln) * ROWSTRIDE];

        // ---- GEMM1: x @ W1, +b1, relu, spill bf16 back to LDS -------------
        v16bf a0 = frag_from_lds(arow, 0, half);
        v16bf a1 = frag_from_lds(arow, 1, half);
#pragma unroll
        for (int nt = 0; nt < 3; ++nt) {
            v8f c = {};
            c = __builtin_amdgcn_wmma_f32_16x16x32_bf16(false, a0, false, w1[nt],     (short)0, c, false, false);
            c = __builtin_amdgcn_wmma_f32_16x16x32_bf16(false, a1, false, w1[3 + nt], (short)0, c, false, false);
#pragma unroll
            for (int r = 0; r < 8; ++r) {
                float v = c[r] + b1v[nt];
                v = v > 0.f ? v : 0.f;
                int m = r + half * 8;   // C layout: row m = r + 8*half, col = ln
                xs16[(rowBase + m) * (2 * ROWSTRIDE) + nt * 16 + ln] = (unsigned short)bf16_rne(v);
            }
        }
        // wave-private LDS region: same-wave DS pipeline is in-order, no barrier.

        // ---- GEMM2: t1 @ W2, +b2, scatter-add to agg[dst] -----------------
        v16bf p0 = frag_from_lds(arow, 0, half);
        v16bf p1 = frag_from_lds(arow, 1, half);
        v8f c2[3];
#pragma unroll
        for (int nt = 0; nt < 3; ++nt) {
            v8f c = {};
            c = __builtin_amdgcn_wmma_f32_16x16x32_bf16(false, p0, false, w2[nt],     (short)0, c, false, false);
            c = __builtin_amdgcn_wmma_f32_16x16x32_bf16(false, p1, false, w2[3 + nt], (short)0, c, false, false);
            c2[nt] = c;
        }
#pragma unroll
        for (int r = 0; r < 8; ++r) {
            int m = r + half * 8;
            int e = eBase + rowBase + m;
            if (e < E) {
                int dn = dst[e];
#pragma unroll
                for (int nt = 0; nt < 3; ++nt)
                    atomAddF(&agg[(size_t)dn * D + nt * 16 + ln], c2[nt][r] + b2v[nt]);
            }
        }
    }
}

// ---------------------------------------------------------------------------
// Node kernel: u = relu(agg@U1+b1)@U2+b2; h += u; agg = 0 (for next layer)
// ---------------------------------------------------------------------------
__global__ __launch_bounds__(THREADS) void node_kernel(
    float* __restrict__ h, float* __restrict__ agg,
    const unsigned* __restrict__ w1f, const float* __restrict__ b1,
    const unsigned* __restrict__ w2f, const float* __restrict__ b2, int N) {

    __shared__ unsigned xs[WG_ROWS * ROWSTRIDE];
    int nBase = blockIdx.x * WG_ROWS;

    {
        int r = threadIdx.x;
        int n = nBase + r;
        uint4* row4 = (uint4*)&xs[r * ROWSTRIDE];
        if (n < N) {
            float4* ap = (float4*)(agg + (size_t)n * D);
#pragma unroll
            for (int j = 0; j < 6; ++j) {
                float4 a0 = ap[2 * j], a1 = ap[2 * j + 1];
                uint4 o;
                o.x = pack2(a0.x, a0.y);
                o.y = pack2(a0.z, a0.w);
                o.z = pack2(a1.x, a1.y);
                o.w = pack2(a1.z, a1.w);
                row4[j] = o;
                ap[2 * j]     = make_float4(0.f, 0.f, 0.f, 0.f);   // reset for next layer
                ap[2 * j + 1] = make_float4(0.f, 0.f, 0.f, 0.f);
            }
        } else {
#pragma unroll
            for (int j = 0; j < 6; ++j) row4[j] = make_uint4(0u, 0u, 0u, 0u);
        }
        row4[6] = make_uint4(0u, 0u, 0u, 0u);
        row4[7] = make_uint4(0u, 0u, 0u, 0u);
    }
    __syncthreads();

    int lane = threadIdx.x & 31;
    int wave = threadIdx.x >> 5;
    int half = lane >> 4, ln = lane & 15;

    v16bf w1[6], w2[6];
#pragma unroll
    for (int f = 0; f < 6; ++f) { w1[f] = load_wfrag(w1f, f, lane); w2[f] = load_wfrag(w2f, f, lane); }
    float b1v[3], b2v[3];
#pragma unroll
    for (int nt = 0; nt < 3; ++nt) { b1v[nt] = b1[nt * 16 + ln]; b2v[nt] = b2[nt * 16 + ln]; }

    unsigned short* xs16 = (unsigned short*)xs;

    for (int t = 0; t < TILES_PER_WAVE; ++t) {
        int rowBase = (wave * TILES_PER_WAVE + t) * 16;
        const unsigned* arow = &xs[(rowBase + ln) * ROWSTRIDE];

        v16bf a0 = frag_from_lds(arow, 0, half);
        v16bf a1 = frag_from_lds(arow, 1, half);
#pragma unroll
        for (int nt = 0; nt < 3; ++nt) {
            v8f c = {};
            c = __builtin_amdgcn_wmma_f32_16x16x32_bf16(false, a0, false, w1[nt],     (short)0, c, false, false);
            c = __builtin_amdgcn_wmma_f32_16x16x32_bf16(false, a1, false, w1[3 + nt], (short)0, c, false, false);
#pragma unroll
            for (int r = 0; r < 8; ++r) {
                float v = c[r] + b1v[nt];
                v = v > 0.f ? v : 0.f;
                int m = r + half * 8;
                xs16[(rowBase + m) * (2 * ROWSTRIDE) + nt * 16 + ln] = (unsigned short)bf16_rne(v);
            }
        }

        v16bf p0 = frag_from_lds(arow, 0, half);
        v16bf p1 = frag_from_lds(arow, 1, half);
        v8f c2[3];
#pragma unroll
        for (int nt = 0; nt < 3; ++nt) {
            v8f c = {};
            c = __builtin_amdgcn_wmma_f32_16x16x32_bf16(false, p0, false, w2[nt],     (short)0, c, false, false);
            c = __builtin_amdgcn_wmma_f32_16x16x32_bf16(false, p1, false, w2[3 + nt], (short)0, c, false, false);
            c2[nt] = c;
        }
#pragma unroll
        for (int r = 0; r < 8; ++r) {
            int m = r + half * 8;
            int n = nBase + rowBase + m;
            if (n < N) {
#pragma unroll
                for (int nt = 0; nt < 3; ++nt) {
                    size_t idx = (size_t)n * D + nt * 16 + ln;   // unique per lane
                    h[idx] += c2[nt][r] + b2v[nt];
                }
            }
        }
    }
}

// ---------------------------------------------------------------------------
// Readout: y = relu(h@ro_w1+b1)@ro_w2+b2 ; out[gid[n]] += y  (tiny: VALU)
// ---------------------------------------------------------------------------
__global__ __launch_bounds__(THREADS) void readout_kernel(
    const float* __restrict__ h, const int* __restrict__ gid,
    const float* __restrict__ w1, const float* __restrict__ b1,
    const float* __restrict__ w2, const float* __restrict__ b2,
    float* __restrict__ out, int N) {

    __shared__ float w1s[D * D];
    __shared__ float b1s[D];
    __shared__ float w2s[D];
    for (int i = threadIdx.x; i < D * D; i += THREADS) w1s[i] = w1[i];
    if (threadIdx.x < D) { b1s[threadIdx.x] = b1[threadIdx.x]; w2s[threadIdx.x] = w2[threadIdx.x]; }
    __syncthreads();

    int n = blockIdx.x * THREADS + threadIdx.x;
    if (n >= N) return;

    float hv[D];
    const float4* hp = (const float4*)(h + (size_t)n * D);
#pragma unroll
    for (int j = 0; j < 12; ++j) {
        float4 v = hp[j];
        hv[4 * j] = v.x; hv[4 * j + 1] = v.y; hv[4 * j + 2] = v.z; hv[4 * j + 3] = v.w;
    }
    float y = b2[0];
    for (int j = 0; j < D; ++j) {
        float acc = b1s[j];
#pragma unroll
        for (int k = 0; k < D; ++k) acc = fmaf(hv[k], w1s[k * D + j], acc);
        y = fmaf(fmaxf(acc, 0.f), w2s[j], y);
    }
    atomAddF(&out[gid[n]], y);
}

// ---------------------------------------------------------------------------
extern "C" void kernel_launch(void* const* d_in, const int* in_sizes, int n_in,
                              void* d_out, int out_size, void* d_ws, size_t ws_size,
                              hipStream_t stream) {
    const int*   AtomicNum = (const int*)d_in[0];
    const int*   Edge      = (const int*)d_in[1];
    const int*   gid       = (const int*)d_in[2];
    const float* emb       = (const float*)d_in[3];
    const float* msg_w1 = (const float*)d_in[4];  const float* msg_b1 = (const float*)d_in[5];
    const float* msg_w2 = (const float*)d_in[6];  const float* msg_b2 = (const float*)d_in[7];
    const float* upd_w1 = (const float*)d_in[8];  const float* upd_b1 = (const float*)d_in[9];
    const float* upd_w2 = (const float*)d_in[10]; const float* upd_b2 = (const float*)d_in[11];
    const float* ro_w1  = (const float*)d_in[12]; const float* ro_b1  = (const float*)d_in[13];
    const float* ro_w2  = (const float*)d_in[14]; const float* ro_b2  = (const float*)d_in[15];
    float* out = (float*)d_out;

    const int N = in_sizes[0];          // 100000
    const int E = in_sizes[1] / 2;      // 1600000
    const int* src = Edge;
    const int* dst = Edge + E;

    // workspace: h | agg | weight fragments (all offsets 256B-aligned)
    char*     ws    = (char*)d_ws;
    size_t    hB    = (size_t)N * D * sizeof(float);    // 19.2 MB
    float*    h     = (float*)ws;
    float*    agg   = (float*)(ws + hB);
    unsigned* frags = (unsigned*)(ws + 2 * hB);         // 24 * 6KB

    zero_out_kernel<<<(out_size + THREADS - 1) / THREADS, THREADS, 0, stream>>>(out, out_size);
    prep_frags<<<24, 192, 0, stream>>>(msg_w1, msg_w2, upd_w1, upd_w2, frags);
    embed_kernel<<<((N * D) + THREADS - 1) / THREADS, THREADS, 0, stream>>>(AtomicNum, emb, h, agg, N * D);

    int edgeBlocks = (E + WG_ROWS - 1) / WG_ROWS;
    int nodeBlocks = (N + WG_ROWS - 1) / WG_ROWS;
    for (int l = 0; l < 6; ++l) {
        const unsigned* mw1f = frags + (size_t)(0 * 6 + l) * 1536;
        const unsigned* mw2f = frags + (size_t)(1 * 6 + l) * 1536;
        const unsigned* uw1f = frags + (size_t)(2 * 6 + l) * 1536;
        const unsigned* uw2f = frags + (size_t)(3 * 6 + l) * 1536;
        edge_kernel<<<edgeBlocks, THREADS, 0, stream>>>(h, agg, src, dst,
                                                        mw1f, msg_b1 + l * D, mw2f, msg_b2 + l * D, E);
        node_kernel<<<nodeBlocks, THREADS, 0, stream>>>(h, agg,
                                                        uw1f, upd_b1 + l * D, uw2f, upd_b2 + l * D, N);
    }
    readout_kernel<<<(N + THREADS - 1) / THREADS, THREADS, 0, stream>>>(h, gid, ro_w1, ro_b1, ro_w2, ro_b2, out, N);
}
